// GPT_13864154431648
// MI455X (gfx1250) — matrix operand order repaired and checked by reading, not compiled, not run on hardware
//
#include <hip/hip_runtime.h>

typedef __attribute__((ext_vector_type(16))) __bf16 v16bf;
typedef __attribute__((ext_vector_type(8)))  __bf16 v8bf;
typedef __attribute__((ext_vector_type(8)))  float  v8f;
typedef __attribute__((ext_vector_type(4)))  unsigned int u32x4;
typedef __attribute__((ext_vector_type(8)))  int i32x8;
typedef __attribute__((ext_vector_type(4)))  int i32x4;

#define TPB 256

#if defined(__has_builtin)
#if __has_builtin(__builtin_amdgcn_tensor_load_to_lds) && __has_builtin(__builtin_amdgcn_s_wait_tensorcnt)
#define HAVE_TDM 1
#endif
#endif
#ifndef HAVE_TDM
#define HAVE_TDM 0
#endif

// ---------------------------------------------------------------- reductions
__device__ __forceinline__ float block_reduce_sum(float v, float* red) {
  int t = threadIdx.x;
  red[t] = v; __syncthreads();
  for (int off = TPB / 2; off > 0; off >>= 1) {
    if (t < off) red[t] += red[t + off];
    __syncthreads();
  }
  float r = red[0];
  __syncthreads();
  return r;
}

__device__ __forceinline__ float block_reduce_max(float v, float* red) {
  int t = threadIdx.x;
  red[t] = v; __syncthreads();
  for (int off = TPB / 2; off > 0; off >>= 1) {
    if (t < off) red[t] = fmaxf(red[t], red[t + off]);
    __syncthreads();
  }
  float r = red[0];
  __syncthreads();
  return r;
}

// ---------------------------------------------------------------- embedding (fp32 residual stream)
__global__ void __launch_bounds__(TPB)
embed_kernel(const int* __restrict__ ids, const float* __restrict__ tok,
             const float* __restrict__ pos, float* __restrict__ x,
             int T, int S, int D) {
  long i = (long)blockIdx.x * TPB + threadIdx.x;
  if (i >= (long)T * D) return;
  int d  = (int)(i % D);
  long t = i / D;
  int sp = (int)(t % S);
  int id = ids[t];
  x[i] = tok[(long)id * D + d] + pos[(long)sp * D + d];
}

// ---------------------------------------------------------------- layernorm: fp32 in -> bf16 out
__global__ void __launch_bounds__(TPB)
ln_kernel(const float* __restrict__ x, const float* __restrict__ g,
          const float* __restrict__ b, __bf16* __restrict__ y, int D) {
  __shared__ float red[TPB];
  long row = blockIdx.x;
  const float* xr = x + row * D;
  float s = 0.0f;
  for (int i = threadIdx.x; i < D; i += TPB) s += xr[i];
  float mu = block_reduce_sum(s, red) / (float)D;
  float vs = 0.0f;
  for (int i = threadIdx.x; i < D; i += TPB) { float d = xr[i] - mu; vs += d * d; }
  float var = block_reduce_sum(vs, red) / (float)D;
  float rstd = rsqrtf(var + 1e-5f);
  __bf16* yr = y + row * D;
  for (int i = threadIdx.x; i < D; i += TPB)
    yr[i] = (__bf16)((xr[i] - mu) * rstd * g[i] + b[i]);
}

// ---------------------------------------------------------------- causal softmax: fp32 scores -> bf16 probs
__global__ void __launch_bounds__(TPB)
softmax_causal_kernel(const float* __restrict__ scores, __bf16* __restrict__ probs, int S) {
  __shared__ float red[TPB];
  long row = blockIdx.x;          // (b*H + h)*S + q
  int q = (int)(row % S);
  const float* sr = scores + row * S;
  __bf16* pr = probs + row * S;
  float mx = -3.0e38f;
  for (int j = threadIdx.x; j < S; j += TPB) {
    float v = (j <= q) ? sr[j] : -3.0e38f;
    mx = fmaxf(mx, v);
  }
  mx = block_reduce_max(mx, red);
  float ev[8];
  float sum = 0.0f;
  int c = 0;
  for (int j = threadIdx.x; j < S; j += TPB) {
    float e = (j <= q) ? __expf(sr[j] - mx) : 0.0f;
    ev[c++] = e;
    sum += e;
  }
  sum = block_reduce_sum(sum, red);
  float inv = 1.0f / sum;
  c = 0;
  for (int j = threadIdx.x; j < S; j += TPB) pr[j] = (__bf16)(ev[c++] * inv);
}

// ---------------------------------------------------------------- fp32 -> bf16 copy
__global__ void __launch_bounds__(TPB)
cvt_bf16_kernel(const float* __restrict__ in, __bf16* __restrict__ out, long n) {
  long i = (long)blockIdx.x * TPB + threadIdx.x;
  if (i < n) out[i] = (__bf16)in[i];
}

// ---------------------------------------------------------------- weight transpose: W[K][N] f32 -> Wt[N][K] bf16
__global__ void __launch_bounds__(TPB)
wt_kernel(const float* __restrict__ W, __bf16* __restrict__ Wt, int K, int N) {
  __shared__ float tile[32][33];
  int n0 = blockIdx.x * 32, k0 = blockIdx.y * 32;
  int tx = threadIdx.x & 31, ty = threadIdx.x >> 5;     // 32 x 8
  for (int r = ty; r < 32; r += 8)
    tile[r][tx] = W[(long)(k0 + r) * N + (n0 + tx)];
  __syncthreads();
  for (int r = ty; r < 32; r += 8)
    Wt[(long)(n0 + r) * K + (k0 + tx)] = (__bf16)tile[tx][r];
}

// ---------------------------------------------------------------- V transpose: qkv bf16 -> vt[z][d][kj] bf16
__global__ void __launch_bounds__(TPB)
vt_kernel(const __bf16* __restrict__ qkv, __bf16* __restrict__ vt,
          int S, int D3, int Dq, int H, int DH, long total) {
  long i = (long)blockIdx.x * TPB + threadIdx.x;
  if (i >= total) return;
  int kj = (int)(i % S);
  long r = i / S;
  int d  = (int)(r % DH);
  int z  = (int)(r / DH);
  int b = z / H, h = z % H;
  vt[i] = qkv[(long)(b * S + kj) * D3 + 2 * Dq + h * DH + d];
}

// ---------------------------------------------------------------- TDM tile load: 2D bf16 tile -> padded LDS rows
#if HAVE_TDM
__device__ __forceinline__ void tdm_load(unsigned lds_byte_addr, const __bf16* g,
                                         unsigned rows_rem, unsigned k_rem,
                                         unsigned tile_rows, long row_stride_elems) {
  unsigned long ga = (unsigned long)g;
  u32x4 g0;
  g0[0] = 1u;                                        // count=1, user mode
  g0[1] = lds_byte_addr;                             // lds_addr
  g0[2] = (unsigned)ga;                              // global_addr[31:0]
  g0[3] = (unsigned)((ga >> 32) & 0x01ffffffu) | (2u << 30);  // global_addr[56:32], type=2
  i32x8 g1;
  // data_size=1 (2B), pad_enable, pad_interval=code3 (16 dwords = 64B row),
  // pad_amount=code3 (4 dwords = 16B) -> LDS row pitch 80B = LDA(40) bf16
  g1[0] = (int)((1u << 16) | (1u << 20) | (3u << 22) | (3u << 25));
  g1[1] = (int)((k_rem & 0xffffu) << 16);            // tensor_dim0[15:0]
  g1[2] = (int)((k_rem >> 16) | ((rows_rem & 0xffffu) << 16));   // dim0 hi | dim1 lo
  g1[3] = (int)((rows_rem >> 16) | (32u << 16));     // dim1 hi | tile_dim0 = 32
  g1[4] = (int)tile_rows;                            // tile_dim1 | tile_dim2=0
  g1[5] = (int)(unsigned)((unsigned long)row_stride_elems & 0xffffffffu);
  g1[6] = (int)(unsigned)((unsigned long)row_stride_elems >> 32);  // stride hi | dim1_stride lo=0
  g1[7] = 0;
  i32x4 z4 = {0, 0, 0, 0};
  i32x8 z8 = {0, 0, 0, 0, 0, 0, 0, 0};
  __builtin_amdgcn_tensor_load_to_lds(g0, g1, z4, z4, z8, 0);
}
#endif

// ---------------------------------------------------------------- generic bf16 WMMA GEMM
// C[m,n] = scale * sum_k A[m,k]*B[n,k]  (+bias) (+gelu) (+resid); A,B bf16; C fp32 or bf16.
// A element (m,k): A + Ab_b*b + Ab_h*h + m*As_m + k     (k contiguous)
// B element (n,k): B + Bb_b*b + Bb_h*h + n*Bs_n + k     (k contiguous; B pre-transposed)
// C element (m,n): C + Cb_b*b + Cb_h*h + m*Cs_m + n
template <int BM, int BN>
__global__ void __launch_bounds__(TPB)
gemm_bf16_wmma(const __bf16* __restrict__ A, const __bf16* __restrict__ B,
               float* __restrict__ Cf, __bf16* __restrict__ Ch,
               const float* __restrict__ bias, const float* __restrict__ resid,
               int M, int N, int K,
               long As_m, long Ab_b, long Ab_h,
               long Bs_n, long Bb_b, long Bb_h,
               long Cs_m, long Cb_b, long Cb_h,
               int nH, float scale, int do_gelu) {
  constexpr int KT  = 32;
  constexpr int LDA = 40;                  // 32 bf16 + 8 pad = 80B pitch
  constexpr int WN  = BN / 32;
  static_assert((BM / 32) * (BN / 32) == 8, "8 waves per block");

  __shared__ __align__(16) __bf16 sA[2][BM * LDA];
  __shared__ __align__(16) __bf16 sB[2][BN * LDA];

  const int z  = blockIdx.z;
  const int bb = z / nH, hh = z % nH;
  const int mBlk = blockIdx.x * BM;
  const int nBlk = blockIdx.y * BN;

  const __bf16* Abase = A + Ab_b * bb + Ab_h * hh + (long)mBlk * As_m;
  const __bf16* Bbase = B + Bb_b * bb + Bb_h * hh + (long)nBlk * Bs_n;
  float*  Cfb = Cf ? (Cf + Cb_b * bb + Cb_h * hh) : nullptr;
  __bf16* Chb = Ch ? (Ch + Cb_b * bb + Cb_h * hh) : nullptr;
  const float* Rb = resid ? (resid + Cb_b * bb + Cb_h * hh) : nullptr;

  const int tid  = threadIdx.x;
  const int wave = tid >> 5;
  const int lane = tid & 31;
  const int lg   = lane >> 4;
  const int lr   = lane & 15;
  const int wm   = wave / WN;
  const int wn   = wave % WN;

  v8f acc[2][2] = {};
  const int nk = K / KT;

#if HAVE_TDM
  const unsigned ldsA[2] = { (unsigned)(unsigned long)(void*)&sA[0][0],
                             (unsigned)(unsigned long)(void*)&sA[1][0] };
  const unsigned ldsB[2] = { (unsigned)(unsigned long)(void*)&sB[0][0],
                             (unsigned)(unsigned long)(void*)&sB[1][0] };
  if (wave == 0) {
    tdm_load(ldsA[0], Abase, (unsigned)(M - mBlk), (unsigned)K, BM, As_m);
    tdm_load(ldsB[0], Bbase, (unsigned)(N - nBlk), (unsigned)K, BN, Bs_n);
  }
#endif

  for (int t = 0; t < nk; ++t) {
#if HAVE_TDM
    const int buf = t & 1;
    if (wave == 0) __builtin_amdgcn_s_wait_tensorcnt(0);
    __syncthreads();                       // tile[buf] ready; tile[buf^1] free
    if (wave == 0 && t + 1 < nk) {
      long ko = (long)(t + 1) * KT;
      tdm_load(ldsA[buf ^ 1], Abase + ko, (unsigned)(M - mBlk), (unsigned)(K - (t + 1) * KT), BM, As_m);
      tdm_load(ldsB[buf ^ 1], Bbase + ko, (unsigned)(N - nBlk), (unsigned)(K - (t + 1) * KT), BN, Bs_n);
    }
#else
    const int buf = 0;
    const int kt = t * KT;
    __syncthreads();
    for (int i = tid; i < BM * 4; i += TPB) {        // BM rows x 4 segs of 8 bf16
      int r = i >> 2, sseg = i & 3;
      v8bf v = *(const v8bf*)(Abase + (long)r * As_m + kt + sseg * 8);
      *(v8bf*)&sA[0][r * LDA + sseg * 8] = v;
    }
    for (int i = tid; i < BN * 4; i += TPB) {
      int r = i >> 2, sseg = i & 3;
      v8bf v = {};
      if (nBlk + r < N) v = *(const v8bf*)(Bbase + (long)r * Bs_n + kt + sseg * 8);
      *(v8bf*)&sB[0][r * LDA + sseg * 8] = v;
    }
    __syncthreads();
#endif

    union Frag { v16bf v; v8bf h[2]; };
    v16bf afrag[2], bfrag[2];
#pragma unroll
    for (int i = 0; i < 2; ++i) {
      const __bf16* base = &sA[buf][(wm * 32 + i * 16 + lr) * LDA];
      Frag f;
      f.h[0] = *(const v8bf*)(base + lg * 8);        // K = lg*8 + 0..7
      f.h[1] = *(const v8bf*)(base + 16 + lg * 8);   // K = 16 + lg*8 + 0..7
      afrag[i] = f.v;
    }
#pragma unroll
    for (int j = 0; j < 2; ++j) {
      const __bf16* base = &sB[buf][(wn * 32 + j * 16 + lr) * LDA];
      Frag f;
      f.h[0] = *(const v8bf*)(base + lg * 8);        // K = lg*8 + 0..7
      f.h[1] = *(const v8bf*)(base + 16 + lg * 8);   // K = 16 + lg*8 + 0..7
      bfrag[j] = f.v;
    }

#pragma unroll
    for (int i = 0; i < 2; ++i)
#pragma unroll
      for (int j = 0; j < 2; ++j)
        acc[i][j] = __builtin_amdgcn_wmma_f32_16x16x32_bf16(
            false, afrag[i], false, bfrag[j], (short)0, acc[i][j], false, false);
  }

  // epilogue: C layout VGPR r -> m = lg*8 + r, n = lr
#pragma unroll
  for (int j = 0; j < 2; ++j) {
    int n = nBlk + wn * 32 + j * 16 + lr;
    if (n >= N) continue;
    float bval = bias ? bias[n] : 0.0f;
#pragma unroll
    for (int i = 0; i < 2; ++i) {
#pragma unroll
      for (int r = 0; r < 8; ++r) {
        int m = mBlk + wm * 32 + i * 16 + lg * 8 + r;
        float v = acc[i][j][r] * scale + bval;
        if (do_gelu) v = 0.5f * v * (1.0f + erff(v * 0.70710678118f));
        long cidx = (long)m * Cs_m + n;
        if (Rb) v += Rb[cidx];
        if (Chb) Chb[cidx] = (__bf16)v;
        else     Cfb[cidx] = v;
      }
    }
  }
}

// ---------------------------------------------------------------- host driver
extern "C" void kernel_launch(void* const* d_in, const int* in_sizes, int n_in,
                              void* d_out, int out_size, void* d_ws, size_t ws_size,
                              hipStream_t stream) {
  constexpr int V = 50257, D = 768, H = 12, F = 3072, L = 4, S = 1024, Bsz = 2;
  constexpr int T = Bsz * S, DH = D / H, D3 = 3 * D;

  const int*   ids   = (const int*)d_in[0];
  const float* tok   = (const float*)d_in[1];
  const float* pos   = (const float*)d_in[2];
  const float* ln1_g = (const float*)d_in[3];
  const float* ln1_b = (const float*)d_in[4];
  const float* qkv_w = (const float*)d_in[5];
  const float* qkv_b = (const float*)d_in[6];
  const float* proj_w= (const float*)d_in[7];
  const float* proj_b= (const float*)d_in[8];
  const float* ln2_g = (const float*)d_in[9];
  const float* ln2_b = (const float*)d_in[10];
  const float* fc1_w = (const float*)d_in[11];
  const float* fc1_b = (const float*)d_in[12];
  const float* fc2_w = (const float*)d_in[13];
  const float* fc2_b = (const float*)d_in[14];
  const float* lnf_g = (const float*)d_in[15];
  const float* lnf_b = (const float*)d_in[16];
  float* out = (float*)d_out;
  (void)in_sizes; (void)n_in; (void)out_size; (void)ws_size;

  // ---- workspace layout (256B aligned slabs)
  char* w = (char*)d_ws;
  auto take = [&](long bytes) { char* p = w; w += (bytes + 255) & ~255L; return p; };
  float*  x       = (float*)take((long)T * D * 4);
  __bf16* lnb     = (__bf16*)take((long)T * D * 2);
  __bf16* qkv     = (__bf16*)take((long)T * D3 * 2);
  __bf16* vt      = (__bf16*)take((long)Bsz * H * DH * S * 2);
  __bf16* attnout = (__bf16*)take((long)T * D * 2);
  __bf16* hfc     = (__bf16*)take((long)T * F * 2);
  __bf16* wT      = (__bf16*)take((long)(D * D3 + D * D + D * F + F * D) * 2);
  __bf16* probs   = (__bf16*)take((long)Bsz * H * S * S * 2);
  float*  scores  = (float*)take((long)Bsz * H * S * S * 4);
  __bf16* tokT    = (__bf16*)scores;     // alias: tok_emb bf16 copy reuses dead scores slab

  __bf16* qkvT = wT;
  __bf16* prjT = qkvT + (long)D3 * D;
  __bf16* fc1T = prjT + (long)D * D;
  __bf16* fc2T = fc1T + (long)F * D;

  // 1) embeddings
  {
    long n = (long)T * D;
    embed_kernel<<<(int)((n + TPB - 1) / TPB), TPB, 0, stream>>>(ids, tok, pos, x, T, S, D);
  }

  for (int l = 0; l < L; ++l) {
    // per-layer weight transposes -> bf16 [N][K]
    wt_kernel<<<dim3(D3 / 32, D / 32), TPB, 0, stream>>>(qkv_w  + (long)l * D * D3, qkvT, D, D3);
    wt_kernel<<<dim3(D / 32,  D / 32), TPB, 0, stream>>>(proj_w + (long)l * D * D,  prjT, D, D);
    wt_kernel<<<dim3(F / 32,  D / 32), TPB, 0, stream>>>(fc1_w  + (long)l * D * F,  fc1T, D, F);
    wt_kernel<<<dim3(D / 32,  F / 32), TPB, 0, stream>>>(fc2_w  + (long)l * F * D,  fc2T, F, D);

    // ln1 -> bf16
    ln_kernel<<<T, TPB, 0, stream>>>(x, ln1_g + (long)l * D, ln1_b + (long)l * D, lnb, D);

    // qkv(bf16) = lnb @ qkv_w + b
    gemm_bf16_wmma<64, 128><<<dim3(T / 64, D3 / 128, 1), TPB, 0, stream>>>(
        lnb, qkvT, nullptr, qkv, qkv_b + (long)l * D3, nullptr,
        T, D3, D,
        D, 0, 0,
        D, 0, 0,
        D3, 0, 0,
        1, 1.0f, 0);

    // vt[z][d][kj] from qkv
    {
      long tot = (long)Bsz * H * DH * S;
      vt_kernel<<<(int)((tot + TPB - 1) / TPB), TPB, 0, stream>>>(qkv, vt, S, D3, D, H, DH, tot);
    }

    // scores(fp32) = Q.K^T / 8
    gemm_bf16_wmma<64, 128><<<dim3(S / 64, S / 128, Bsz * H), TPB, 0, stream>>>(
        qkv /*Q*/, qkv + D /*K rows*/, scores, nullptr, nullptr, nullptr,
        S, S, DH,
        D3, (long)S * D3, DH,
        D3, (long)S * D3, DH,
        S, (long)H * S * S, (long)S * S,
        H, 0.125f, 0);

    // probs(bf16) = causal softmax(scores)
    softmax_causal_kernel<<<Bsz * H * S, TPB, 0, stream>>>(scores, probs, S);

    // attnout(bf16)[:, h*DH:(h+1)*DH] = P @ V  (B = vt rows are d, k is kj)
    gemm_bf16_wmma<128, 64><<<dim3(S / 128, 1, Bsz * H), TPB, 0, stream>>>(
        probs, vt, nullptr, attnout, nullptr, nullptr,
        S, DH, S,
        S, (long)H * S * S, (long)S * S,
        S, (long)H * DH * S, (long)DH * S,
        D, (long)S * D, DH,
        H, 1.0f, 0);

    // x(fp32) = attnout @ proj_w + b + x
    gemm_bf16_wmma<64, 128><<<dim3(T / 64, D / 128, 1), TPB, 0, stream>>>(
        attnout, prjT, x, nullptr, proj_b + (long)l * D, x,
        T, D, D,
        D, 0, 0,
        D, 0, 0,
        D, 0, 0,
        1, 1.0f, 0);

    // ln2 -> bf16
    ln_kernel<<<T, TPB, 0, stream>>>(x, ln2_g + (long)l * D, ln2_b + (long)l * D, lnb, D);

    // hfc(bf16) = gelu(lnb @ fc1_w + b)
    gemm_bf16_wmma<64, 128><<<dim3(T / 64, F / 128, 1), TPB, 0, stream>>>(
        lnb, fc1T, nullptr, hfc, fc1_b + (long)l * F, nullptr,
        T, F, D,
        D, 0, 0,
        D, 0, 0,
        F, 0, 0,
        1, 1.0f, 1);

    // x(fp32) = hfc @ fc2_w + b + x
    gemm_bf16_wmma<64, 128><<<dim3(T / 64, D / 128, 1), TPB, 0, stream>>>(
        hfc, fc2T, x, nullptr, fc2_b + (long)l * D, x,
        T, D, F,
        F, 0, 0,
        F, 0, 0,
        D, 0, 0,
        1, 1.0f, 0);
  }

  // final LN -> bf16
  ln_kernel<<<T, TPB, 0, stream>>>(x, lnf_g, lnf_b, lnb, D);

  // tok_emb -> bf16 (aliases dead scores slab); already [V][D] = [N][K]
  {
    long n = (long)V * D;
    cvt_bf16_kernel<<<(int)((n + TPB - 1) / TPB), TPB, 0, stream>>>(tok, tokT, n);
  }

  // logits(fp32) = lnb @ tok_emb^T
  gemm_bf16_wmma<64, 128><<<dim3(T / 64, (V + 127) / 128, 1), TPB, 0, stream>>>(
      lnb, tokT, out, nullptr, nullptr, nullptr,
      T, V, D,
      D, 0, 0,
      D, 0, 0,
      V, 0, 0,
      1, 1.0f, 0);
}
